// TokenRoutedMLP_35321811042576
// MI455X (gfx1250) — matrix operand order repaired and checked by reading, not compiled, not run on hardware
//
#include <hip/hip_runtime.h>
#include <hip/hip_bf16.h>

// ---------------------------------------------------------------------------
// CDNA5 (gfx1250, wave32) token-routed MoE SwiGLU using v_wmma_f32_16x16x32_f16
// Software-pipelined: global->reg (next tile) overlaps WMMA on current tile.
// Down-projection A tiles use GLOBAL_LOAD_ASYNC_TO_LDS (ASYNCcnt) + double buf.
// ---------------------------------------------------------------------------

typedef __attribute__((ext_vector_type(16))) _Float16 v16h;
typedef __attribute__((ext_vector_type(8)))  float    v8f;

union Frag16 {           // one WMMA 16-bit operand: 16 halves = 8 VGPRs
    v16h v;
    unsigned u[8];
};

union H4 {               // 4 halves <-> one 8-byte LDS store
    _Float16 h[4];
    uint2 u;
};

#define NUM_EXPERTS 8
#define TM 32            // tokens per block tile
#define TN 128           // output cols per block tile
#define TK 32            // K step (matches 16x16x32 WMMA)
#define AP 36            // LDS pitch for A rows (K padded, multiple of 4)
#define BP 36            // LDS pitch for B cols (K padded, multiple of 4)

// ---------------------------------------------------------------------------
// Routing: expert = clip(token_id) % 8 ; counting-gather into per-expert lists
// ---------------------------------------------------------------------------
__global__ void zero_counts_kernel(int* counts) {
    if (threadIdx.x < NUM_EXPERTS) counts[threadIdx.x] = 0;
}

__global__ void route_kernel(const int* __restrict__ ids, int* __restrict__ counts,
                             int* __restrict__ perm, int n, int vocab) {
    int i = blockIdx.x * blockDim.x + threadIdx.x;
    if (i >= n) return;
    int t = ids[i];
    t = t < 0 ? 0 : (t > vocab - 1 ? vocab - 1 : t);
    int e = t & (NUM_EXPERTS - 1);           // % 8
    int pos = atomicAdd(&counts[e], 1);
    perm[e * n + pos] = i;
}

// ---------------------------------------------------------------------------
// Fragment loaders (ISA 7.12.2 16-bit operand layouts), 8 dword LDS reads each
// ---------------------------------------------------------------------------
__device__ __forceinline__ void load_a_frag(Frag16& a, const _Float16* aL,
                                            int mrow, int half) {
    #pragma unroll
    for (int j = 0; j < 8; j++) {
        int i2 = 2 * j;
        int k = i2 + 8 * half + ((i2 >= 8) ? 8 : 0);
        a.u[j] = *(const unsigned*)&aL[mrow * AP + k];
    }
}
__device__ __forceinline__ void load_b_frag(Frag16& b, const _Float16* bL,
                                            int ncol, int half) {
    #pragma unroll
    for (int j = 0; j < 8; j++) {
        int kk = 2 * j + 8 * half + ((2 * j >= 8) ? 8 : 0);
        b.u[j] = *(const unsigned*)&bL[ncol * BP + kk];
    }
}

__device__ __forceinline__ uint2 cvt4(float4 v) {
    H4 t;
    t.h[0] = (_Float16)v.x; t.h[1] = (_Float16)v.y;
    t.h[2] = (_Float16)v.z; t.h[3] = (_Float16)v.w;
    return t.u;
}

// ---- CDNA5 async global->LDS copy (8 bytes per lane, tracked by ASYNCcnt)
__device__ __forceinline__ void async_lds_b64(unsigned lds_byte_off, const void* gaddr) {
    asm volatile("global_load_async_to_lds_b64 %0, %1, off"
                 :: "v"(lds_byte_off), "v"(gaddr) : "memory");
}
__device__ __forceinline__ void wait_async_le1() {
    asm volatile("s_wait_asynccnt 0x1" ::: "memory");
}
__device__ __forceinline__ void wait_async_le0() {
    asm volatile("s_wait_asynccnt 0x0" ::: "memory");
}

// ---------------------------------------------------------------------------
// Fused gate/up GEMM + SiLU epilogue.
//   inter[token, nt] = silu(X @ Wg) * (X @ Wu)   (f16 output)
// BT=0: W is [K, N] row-major, ld ldb (routed experts)
// BT=1: W is [N, K] row-major, ld ldb (shared expert, x @ W.T)
// ROUTED: A rows gathered through per-expert perm list (length counts[e]).
// ---------------------------------------------------------------------------
template<bool BT, bool ROUTED>
__global__ __launch_bounds__(256) void gateup_kernel(
    const float* __restrict__ X, const float* __restrict__ Wg,
    const float* __restrict__ Wu, _Float16* __restrict__ inter,
    const int* __restrict__ perm, const int* __restrict__ counts,
    int ntok, int K, int Ncols, int ldb, long long wstride)
{
    const int e = blockIdx.z;
    const int n_rows = ROUTED ? counts[e] : ntok;
    const int mt0 = blockIdx.x * TM;
    if (mt0 >= n_rows) return;
    const int nt0 = blockIdx.y * TN;
    const int* pm = ROUTED ? perm + (long long)e * ntok : nullptr;
    const float* wg = ROUTED ? Wg + (long long)e * wstride : Wg;
    const float* wu = ROUTED ? Wu + (long long)e * wstride : Wu;

    __shared__ __align__(16) _Float16 aL[TM * AP];
    __shared__ __align__(16) _Float16 bgL[TN * BP];
    __shared__ __align__(16) _Float16 buL[TN * BP];

    const int tid  = threadIdx.x;
    const int lane = tid & 31;
    const int wv   = tid >> 5;          // 0..7
    const int m0   = (wv & 1) * 16;     // wave M offset in tile
    const int n0   = (wv >> 1) * 32;    // wave N offset in tile
    const int half = lane >> 4;
    const int ml   = lane & 15;

    // ---- hoisted A staging addresses: one float4 (row segment) per thread
    const int ar = tid >> 3;            // 0..31 (row in tile)
    const int ak = (tid & 7) * 4;       // 0..28 (k within step)
    int arow = mt0 + ar;
    if (arow > n_rows - 1) arow = n_rows - 1;       // clamp: value never stored
    const int atok = ROUTED ? pm[arow] : arow;
    unsigned aoff = (unsigned)atok * (unsigned)K + ak;

    // ---- hoisted B staging addresses: 4 float4 per thread per matrix
    unsigned boff[4];
    int bk[4], bn[4];
    #pragma unroll
    for (int i = 0; i < 4; i++) {
        int idx = i * 256 + tid;
        if (BT) { bn[i] = idx >> 3; bk[i] = (idx & 7) * 4;
                  boff[i] = (unsigned)(nt0 + bn[i]) * ldb + bk[i]; }
        else    { bk[i] = idx >> 5; bn[i] = (idx & 31) * 4;
                  boff[i] = (unsigned)bk[i] * ldb + nt0 + bn[i]; }
    }
    const unsigned bstep = BT ? (unsigned)TK : (unsigned)(TK * ldb);

    // ---- preload tile 0 into registers
    float4 aReg = *(const float4*)(X + aoff);  aoff += TK;
    float4 gReg[4], uReg[4];
    #pragma unroll
    for (int i = 0; i < 4; i++) {
        gReg[i] = *(const float4*)(wg + boff[i]);
        uReg[i] = *(const float4*)(wu + boff[i]);
        boff[i] += bstep;
    }

    v8f ag0 = {}, ag1 = {}, au0 = {}, au1 = {};
    const int niter = K / TK;

    for (int it = 0; it < niter; ++it) {
        if (it) __syncthreads();        // prior tile's WMMAs done before overwrite
        // ---- commit in-register tile to LDS (fp32 -> f16)
        *(uint2*)&aL[ar * AP + ak] = cvt4(aReg);
        #pragma unroll
        for (int i = 0; i < 4; i++) {
            if (BT) {   // contiguous K in LDS column -> 8B stores
                *(uint2*)&bgL[bn[i] * BP + bk[i]] = cvt4(gReg[i]);
                *(uint2*)&buL[bn[i] * BP + bk[i]] = cvt4(uReg[i]);
            } else {    // 4 consecutive n -> strided b16 stores
                bgL[(bn[i] + 0) * BP + bk[i]] = (_Float16)gReg[i].x;
                bgL[(bn[i] + 1) * BP + bk[i]] = (_Float16)gReg[i].y;
                bgL[(bn[i] + 2) * BP + bk[i]] = (_Float16)gReg[i].z;
                bgL[(bn[i] + 3) * BP + bk[i]] = (_Float16)gReg[i].w;
                buL[(bn[i] + 0) * BP + bk[i]] = (_Float16)uReg[i].x;
                buL[(bn[i] + 1) * BP + bk[i]] = (_Float16)uReg[i].y;
                buL[(bn[i] + 2) * BP + bk[i]] = (_Float16)uReg[i].z;
                buL[(bn[i] + 3) * BP + bk[i]] = (_Float16)uReg[i].w;
            }
        }
        // ---- issue next tile's global loads (latency hidden behind WMMAs)
        if (it + 1 < niter) {
            aReg = *(const float4*)(X + aoff);  aoff += TK;
            #pragma unroll
            for (int i = 0; i < 4; i++) {
                gReg[i] = *(const float4*)(wg + boff[i]);
                uReg[i] = *(const float4*)(wu + boff[i]);
                boff[i] += bstep;
            }
            __builtin_prefetch(wg + boff[0], 0, 1);   // tile it+2
            __builtin_prefetch(wu + boff[0], 0, 1);
        }
        __syncthreads();

        Frag16 a, bg0, bg1, bu0, bu1;
        load_a_frag(a, aL, m0 + ml, half);
        load_b_frag(bg0, bgL, n0 + ml, half);
        load_b_frag(bg1, bgL, n0 + 16 + ml, half);
        load_b_frag(bu0, buL, n0 + ml, half);
        load_b_frag(bu1, buL, n0 + 16 + ml, half);

        ag0 = __builtin_amdgcn_wmma_f32_16x16x32_f16(false, a.v, false, bg0.v, (short)0, ag0, false, false);
        ag1 = __builtin_amdgcn_wmma_f32_16x16x32_f16(false, a.v, false, bg1.v, (short)0, ag1, false, false);
        au0 = __builtin_amdgcn_wmma_f32_16x16x32_f16(false, a.v, false, bu0.v, (short)0, au0, false, false);
        au1 = __builtin_amdgcn_wmma_f32_16x16x32_f16(false, a.v, false, bu1.v, (short)0, au1, false, false);
    }

    // ---- epilogue: silu(gate) * up -> f16 intermediate (D: m=r+8*half, n=ml)
    #pragma unroll
    for (int r = 0; r < 8; r++) {
        int row = mt0 + m0 + r + 8 * half;
        if (row < n_rows) {
            int tok = ROUTED ? pm[row] : row;
            float g0 = ag0[r], u0 = au0[r];
            float g1 = ag1[r], u1 = au1[r];
            float h0 = (g0 / (1.f + __expf(-g0))) * u0;
            float h1 = (g1 / (1.f + __expf(-g1))) * u1;
            unsigned base = (unsigned)tok * (unsigned)Ncols + nt0 + n0;
            inter[base + ml]      = (_Float16)h0;
            inter[base + 16 + ml] = (_Float16)h1;
        }
    }
}

// ---------------------------------------------------------------------------
// Down projection GEMM: out[token, n] (+)= inter[token, :] @ Wd
// A tiles (already f16) copied with async global->LDS DMA, double-buffered.
// ---------------------------------------------------------------------------
template<bool BT, bool ROUTED, bool ACC>
__global__ __launch_bounds__(256) void down_kernel(
    const _Float16* __restrict__ inter, const float* __restrict__ Wd,
    float* __restrict__ out, const int* __restrict__ perm,
    const int* __restrict__ counts, int ntok, int K, int Ncols,
    int ldb, long long wstride)
{
    const int e = blockIdx.z;
    const int n_rows = ROUTED ? counts[e] : ntok;
    const int mt0 = blockIdx.x * TM;
    if (mt0 >= n_rows) return;
    const int nt0 = blockIdx.y * TN;
    const int* pm = ROUTED ? perm + (long long)e * ntok : nullptr;
    const float* wd = ROUTED ? Wd + (long long)e * wstride : Wd;

    __shared__ __align__(16) _Float16 aLb[2][TM * AP];   // double-buffered (async)
    __shared__ __align__(16) _Float16 bL[TN * BP];

    const int tid  = threadIdx.x;
    const int lane = tid & 31;
    const int wv   = tid >> 5;
    const int m0   = (wv & 1) * 16;
    const int n0   = (wv >> 1) * 32;
    const int half = lane >> 4;
    const int ml   = lane & 15;

    // ---- hoisted A staging (source f16): one async b64 per thread per step
    const int ar = tid >> 3;
    const int ak = (tid & 7) * 4;
    int arow = mt0 + ar;
    if (arow > n_rows - 1) arow = n_rows - 1;
    const int atok = ROUTED ? pm[arow] : arow;
    unsigned aoff = (unsigned)atok * (unsigned)K + ak;
    // wave-relative LDS byte offsets of this thread's A slot in each buffer
    const unsigned aslot = (unsigned)(ar * AP + ak) * 2u;
    const unsigned abase0 = (unsigned)(size_t)&aLb[0][0];
    const unsigned abase1 = (unsigned)(size_t)&aLb[1][0];

    // ---- hoisted B staging addresses
    unsigned boff[4];
    int bk[4], bn[4];
    #pragma unroll
    for (int i = 0; i < 4; i++) {
        int idx = i * 256 + tid;
        if (BT) { bn[i] = idx >> 3; bk[i] = (idx & 7) * 4;
                  boff[i] = (unsigned)(nt0 + bn[i]) * ldb + bk[i]; }
        else    { bk[i] = idx >> 5; bn[i] = (idx & 31) * 4;
                  boff[i] = (unsigned)bk[i] * ldb + nt0 + bn[i]; }
    }
    const unsigned bstep = BT ? (unsigned)TK : (unsigned)(TK * ldb);

    // ---- prologue: async A tile 0 -> buf0 ; B tile 0 -> regs
    async_lds_b64(abase0 + aslot, inter + aoff);
    aoff += TK;
    float4 bReg[4];
    #pragma unroll
    for (int i = 0; i < 4; i++) {
        bReg[i] = *(const float4*)(wd + boff[i]);
        boff[i] += bstep;
    }

    v8f ac0 = {}, ac1 = {};
    const int niter = K / TK;
    int cur = 0;

    for (int it = 0; it < niter; ++it) {
        if (it) __syncthreads();   // prior tile's reads done before LDS overwrite
        // ---- commit B tile (fp32 -> f16)
        #pragma unroll
        for (int i = 0; i < 4; i++) {
            if (BT) {
                *(uint2*)&bL[bn[i] * BP + bk[i]] = cvt4(bReg[i]);
            } else {
                bL[(bn[i] + 0) * BP + bk[i]] = (_Float16)bReg[i].x;
                bL[(bn[i] + 1) * BP + bk[i]] = (_Float16)bReg[i].y;
                bL[(bn[i] + 2) * BP + bk[i]] = (_Float16)bReg[i].z;
                bL[(bn[i] + 3) * BP + bk[i]] = (_Float16)bReg[i].w;
            }
        }
        // ---- issue next tile: async A -> alternate buffer, B -> regs
        if (it + 1 < niter) {
            async_lds_b64((cur ? abase0 : abase1) + aslot, inter + aoff);
            aoff += TK;
            #pragma unroll
            for (int i = 0; i < 4; i++) {
                bReg[i] = *(const float4*)(wd + boff[i]);
                boff[i] += bstep;
            }
            __builtin_prefetch(wd + boff[0], 0, 1);
            wait_async_le1();      // tile 'it' landed; tile it+1 may be in flight
        } else {
            wait_async_le0();
        }
        __syncthreads();

        const _Float16* aL = &aLb[cur][0];
        Frag16 a, b0, b1;
        load_a_frag(a, aL, m0 + ml, half);
        load_b_frag(b0, bL, n0 + ml, half);
        load_b_frag(b1, bL, n0 + 16 + ml, half);
        ac0 = __builtin_amdgcn_wmma_f32_16x16x32_f16(false, a.v, false, b0.v, (short)0, ac0, false, false);
        ac1 = __builtin_amdgcn_wmma_f32_16x16x32_f16(false, a.v, false, b1.v, (short)0, ac1, false, false);
        cur ^= 1;
    }

    #pragma unroll
    for (int r = 0; r < 8; r++) {
        int row = mt0 + m0 + r + 8 * half;
        if (row < n_rows) {
            int tok = ROUTED ? pm[row] : row;
            unsigned base = (unsigned)tok * (unsigned)Ncols + nt0 + n0;
            if (ACC) {
                out[base + ml]      += ac0[r];
                out[base + 16 + ml] += ac1[r];
            } else {
                out[base + ml]      = ac0[r];
                out[base + 16 + ml] = ac1[r];
            }
        }
    }
}

// ---------------------------------------------------------------------------
extern "C" void kernel_launch(void* const* d_in, const int* in_sizes, int n_in,
                              void* d_out, int out_size, void* d_ws, size_t ws_size,
                              hipStream_t stream) {
    const float* x      = (const float*)d_in[0];   // [B,S,H]
    const int*   ids    = (const int*)d_in[1];     // [B,S]
    const float* gate_w = (const float*)d_in[2];   // [E,H,IE]
    const float* up_w   = (const float*)d_in[3];   // [E,H,IE]
    const float* down_w = (const float*)d_in[4];   // [E,IE,H]
    const float* sgw    = (const float*)d_in[5];   // [IE,H]
    const float* suw    = (const float*)d_in[6];   // [IE,H]
    const float* sdw    = (const float*)d_in[7];   // [H,IE]
    float* out = (float*)d_out;

    const int H = 1024, IE = 512, VOCAB = 100000;
    const int N = in_sizes[1];                     // B*S tokens

    char* ws = (char*)d_ws;
    int* counts = (int*)ws;                                   // 8 ints
    int* perm   = (int*)(ws + 512);                           // [E, N] ints
    _Float16* inter = (_Float16*)(ws + (size_t)(2u << 20));   // [N, IE] f16

    zero_counts_kernel<<<1, 32, 0, stream>>>(counts);
    route_kernel<<<(N + 255) / 256, 256, 0, stream>>>(ids, counts, perm, N, VOCAB);

    dim3 blk(256);
    // Shared expert: gate/up (B transposed, ldb=H), then down stores d_out.
    dim3 g1((N + TM - 1) / TM, IE / TN, 1);
    gateup_kernel<true, false><<<g1, blk, 0, stream>>>(
        x, sgw, suw, inter, nullptr, nullptr, N, H, IE, /*ldb=*/H, 0);
    dim3 g2((N + TM - 1) / TM, H / TN, 1);
    down_kernel<true, false, false><<<g2, blk, 0, stream>>>(
        inter, sdw, out, nullptr, nullptr, N, IE, H, /*ldb=*/IE, 0);
    // Routed experts: gathered rows per expert list; down accumulates.
    dim3 g3((N + TM - 1) / TM, IE / TN, NUM_EXPERTS);
    gateup_kernel<false, true><<<g3, blk, 0, stream>>>(
        x, gate_w, up_w, inter, perm, counts, N, H, IE, /*ldb=*/IE,
        (long long)H * IE);
    dim3 g4((N + TM - 1) / TM, H / TN, NUM_EXPERTS);
    down_kernel<false, true, true><<<g4, blk, 0, stream>>>(
        inter, down_w, out, perm, counts, N, IE, H, /*ldb=*/H,
        (long long)IE * H);
}
